// FocalReadout_36850819399728
// MI455X (gfx1250) — matrix operand order repaired and checked by reading, not compiled, not run on hardware
//
#include <hip/hip_runtime.h>
#include <hip/hip_bf16.h>

typedef __attribute__((ext_vector_type(16))) _Float16 v16h;
typedef __attribute__((ext_vector_type(8)))  float    v8f;

#define LDS_PITCH 260   // 260 % 64 == 4 -> halves land on disjoint banks

__device__ __forceinline__ float gelu_exact(float v) {
    return 0.5f * v * (1.0f + erff(v * 0.70710678118654752440f));
}

// ---------------------------------------------------------------------------
// Pack fp32 weight matrices [K=256][N=256] (row-major) into WMMA B-operand
// layout, f16. Tiles ordered [nt][kt], each tile = 32 lanes x 16 contiguous
// f16 (lane: n = nt*16 + (lane&15), h = lane>>4; element j -> k = kt*32+h*8+j for j<8, kt*32+16+h*8+(j-8) for j>=8)
// ---------------------------------------------------------------------------
__global__ __launch_bounds__(64) void fr_pack_kernel(
    const float* __restrict__ wi1, const float* __restrict__ wf1,
    const float* __restrict__ wf2,
    _Float16* __restrict__ wi1p, _Float16* __restrict__ wf1p,
    _Float16* __restrict__ wf2p)
{
    int idx = blockIdx.x * 64 + threadIdx.x;   // 3 * 16 * 8 * 32 = 12288 threads
    int matid = idx >> 12;
    int r     = idx & 4095;
    int nt    = r >> 8;
    int kt    = (r >> 5) & 7;
    int lane  = r & 31;
    const float* w = (matid == 0) ? wi1 : (matid == 1) ? wf1 : wf2;
    _Float16*    o = (matid == 0) ? wi1p : (matid == 1) ? wf1p : wf2p;
    int n = nt * 16 + (lane & 15);
    int h = lane >> 4;
    size_t base = ((size_t)(nt * 8 + kt) * 32 + lane) * 16;
#pragma unroll
    for (int j = 0; j < 16; ++j) {
        int k = kt * 32 + (j < 8 ? h * 8 + j : 16 + h * 8 + (j - 8));
        o[base + j] = (_Float16)w[(size_t)k * 256 + n];
    }
}

// ---------------------------------------------------------------------------
// Importance pass: per 16-node wave tile, h = gelu(LN(x@wi1 + bi1)),
// imp = h . wi2 + bi2   -> imp[N]
// ---------------------------------------------------------------------------
__global__ __launch_bounds__(64) void fr_imp_kernel(
    const float* __restrict__ x, const _Float16* __restrict__ w1p,
    const float* __restrict__ bi1, const float* __restrict__ gi,
    const float* __restrict__ bti, const float* __restrict__ wi2,
    const float* __restrict__ bi2, float* __restrict__ imp, int N)
{
    __shared__ float hbuf[2][16][LDS_PITCH];
    const int wave = threadIdx.x >> 5, lane = threadIdx.x & 31;
    const int n = lane & 15, h = lane >> 4;
    const long row0 = ((long)blockIdx.x * 2 + wave) * 16;
    const long row  = row0 + n;
    const bool rv   = row < N;
    const float* xr = x + (size_t)(rv ? row : 0) * 256;

    // A operand: 16-bit A 16x32 layout; lane (m=n, half h) holds
    // k in [kt*32+h*8, +8) and [kt*32+16+h*8, +8) as 16 packed f16.
    v16h a[8];
#pragma unroll
    for (int kt = 0; kt < 8; ++kt) {
        float arr[16];
#pragma unroll
        for (int j = 0; j < 8; ++j) arr[j]     = rv ? xr[kt * 32 + h * 8 + j] : 0.0f;
#pragma unroll
        for (int j = 0; j < 8; ++j) arr[8 + j] = rv ? xr[kt * 32 + 16 + h * 8 + j] : 0.0f;
#pragma unroll
        for (int j = 0; j < 16; ++j) a[kt][j] = (_Float16)arr[j];
    }

    for (int nt = 0; nt < 16; ++nt) {
        v8f c = {};
        const _Float16* bp = w1p + ((size_t)nt * 8 * 32 + lane) * 16;
#pragma unroll
        for (int kt = 0; kt < 8; ++kt) {
            v16h b = *(const v16h*)(bp + (size_t)kt * 512);
            c = __builtin_amdgcn_wmma_f32_16x16x32_f16(false, a[kt], false, b,
                                                       (short)0, c, false, false);
        }
        const float bias = bi1[nt * 16 + n];
#pragma unroll
        for (int j = 0; j < 8; ++j)
            hbuf[wave][h * 8 + j][nt * 16 + n] = c[j] + bias;  // C layout: M=h*8+j, N=lane&15
    }
    __builtin_amdgcn_wave_barrier();

    // LayerNorm over 256 cols of row n: lane pair (n, n+16) split cols
    float s = 0.0f, ss = 0.0f;
    for (int cidx = h * 128; cidx < h * 128 + 128; ++cidx) {
        float v = hbuf[wave][n][cidx]; s += v; ss += v * v;
    }
    s  += __shfl_xor(s, 16, 32);
    ss += __shfl_xor(ss, 16, 32);
    const float mean = s * (1.0f / 256.0f);
    const float var  = ss * (1.0f / 256.0f) - mean * mean;
    const float rstd = rsqrtf(var + 1e-5f);
    float acc = 0.0f;
    for (int cidx = h * 128; cidx < h * 128 + 128; ++cidx) {
        float v = (hbuf[wave][n][cidx] - mean) * rstd * gi[cidx] + bti[cidx];
        acc += gelu_exact(v) * wi2[cidx];
    }
    acc += __shfl_xor(acc, 16, 32);
    if (h == 0 && rv) imp[row] = acc + bi2[0];
}

// ---------------------------------------------------------------------------
// Readout: one workgroup per graph. Segment softmax locally, then fused
// f = gelu(LN(x@wf1+bf1)), t = f@wf2, out[g] = sum_m softmax_w[m]*t[m] + bf2.
// ---------------------------------------------------------------------------
__global__ __launch_bounds__(64) void fr_readout_kernel(
    const float* __restrict__ x, const int* __restrict__ batch,
    const _Float16* __restrict__ wf1p, const _Float16* __restrict__ wf2p,
    const float* __restrict__ bf1, const float* __restrict__ gf,
    const float* __restrict__ btf, const float* __restrict__ bf2,
    const float* __restrict__ imp, float* __restrict__ out, int N)
{
    __shared__ float    hbuf[2][16][LDS_PITCH];
    __shared__ _Float16 fbuf[2][16][256];
    __shared__ float    wrow[2][16];
    __shared__ float    outacc[256];
    __shared__ float    red[64];
    __shared__ float    stats[2];

    const int g = blockIdx.x;
    const int t = threadIdx.x, wave = t >> 5, lane = t & 31;
    const int n = lane & 15, h = lane >> 4;

    // node range of this graph (batch is sorted)
    int lo = 0, hi = N;
    while (lo < hi) { int mid = (lo + hi) >> 1; if (batch[mid] < g) lo = mid + 1; else hi = mid; }
    const int start = lo;
    hi = N;
    while (lo < hi) { int mid = (lo + hi) >> 1; if (batch[mid] < g + 1) lo = mid + 1; else hi = mid; }
    const int end = lo;
    const int cnt = end - start;

    for (int d = t; d < 256; d += 64) outacc[d] = 0.0f;

    // segment max / sum(exp)
    float mx = -1e30f;
    for (int i = start + t; i < end; i += 64) mx = fmaxf(mx, imp[i]);
    red[t] = mx;
    __syncthreads();
    if (t == 0) { float m = red[0]; for (int i = 1; i < 64; ++i) m = fmaxf(m, red[i]); stats[0] = m; }
    __syncthreads();
    mx = stats[0];
    float se = 0.0f;
    for (int i = start + t; i < end; i += 64) se += expf(imp[i] - mx);
    red[t] = se;
    __syncthreads();
    if (t == 0) { float sum = red[0]; for (int i = 1; i < 64; ++i) sum += red[i];
                  stats[1] = (sum > 0.0f) ? 1.0f / sum : 0.0f; }
    __syncthreads();
    const float inv_s = stats[1];

    const int ntiles = (cnt + 15) >> 4;
    for (int tile = wave; tile < ntiles; tile += 2) {   // per-wave, no block barriers inside
        const int  r0  = start + tile * 16;
        const int  row = r0 + n;
        const bool rv  = row < end;
        const float* xr = x + (size_t)(rv ? row : 0) * 256;

        v16h a[8];
#pragma unroll
        for (int kt = 0; kt < 8; ++kt) {
            float arr[16];
#pragma unroll
            for (int j = 0; j < 8; ++j) arr[j]     = rv ? xr[kt * 32 + h * 8 + j] : 0.0f;
#pragma unroll
            for (int j = 0; j < 8; ++j) arr[8 + j] = rv ? xr[kt * 32 + 16 + h * 8 + j] : 0.0f;
#pragma unroll
            for (int j = 0; j < 16; ++j) a[kt][j] = (_Float16)arr[j];
        }

        // GEMM1: x @ wf1 (+bf1) -> hbuf
        for (int nt = 0; nt < 16; ++nt) {
            v8f c = {};
            const _Float16* bp = wf1p + ((size_t)nt * 8 * 32 + lane) * 16;
#pragma unroll
            for (int kt = 0; kt < 8; ++kt) {
                v16h b = *(const v16h*)(bp + (size_t)kt * 512);
                c = __builtin_amdgcn_wmma_f32_16x16x32_f16(false, a[kt], false, b,
                                                           (short)0, c, false, false);
            }
            const float bias = bf1[nt * 16 + n];
#pragma unroll
            for (int j = 0; j < 8; ++j)
                hbuf[wave][h * 8 + j][nt * 16 + n] = c[j] + bias;
        }
        if (h == 0) wrow[wave][n] = rv ? expf(imp[row] - mx) * inv_s : 0.0f;
        __builtin_amdgcn_wave_barrier();

        // LayerNorm + GELU -> f16 fbuf
        float s = 0.0f, ss = 0.0f;
        for (int cidx = h * 128; cidx < h * 128 + 128; ++cidx) {
            float v = hbuf[wave][n][cidx]; s += v; ss += v * v;
        }
        s  += __shfl_xor(s, 16, 32);
        ss += __shfl_xor(ss, 16, 32);
        const float mean = s * (1.0f / 256.0f);
        const float var  = ss * (1.0f / 256.0f) - mean * mean;
        const float rstd = rsqrtf(var + 1e-5f);
        for (int cidx = h * 128; cidx < h * 128 + 128; ++cidx) {
            float v = (hbuf[wave][n][cidx] - mean) * rstd * gf[cidx] + btf[cidx];
            fbuf[wave][n][cidx] = (_Float16)gelu_exact(v);
        }
        __builtin_amdgcn_wave_barrier();

        // reload A (f) from LDS in WMMA A layout
#pragma unroll
        for (int kt = 0; kt < 8; ++kt) {
#pragma unroll
            for (int j = 0; j < 8; ++j) {
                a[kt][j]     = fbuf[wave][n][kt * 32 + h * 8 + j];
                a[kt][8 + j] = fbuf[wave][n][kt * 32 + 16 + h * 8 + j];
            }
        }
        float wv[8];
#pragma unroll
        for (int j = 0; j < 8; ++j) wv[j] = wrow[wave][h * 8 + j];

        // GEMM2: f @ wf2, weighted column-sum into outacc
        for (int nt = 0; nt < 16; ++nt) {
            v8f c = {};
            const _Float16* bp = wf2p + ((size_t)nt * 8 * 32 + lane) * 16;
#pragma unroll
            for (int kt = 0; kt < 8; ++kt) {
                v16h b = *(const v16h*)(bp + (size_t)kt * 512);
                c = __builtin_amdgcn_wmma_f32_16x16x32_f16(false, a[kt], false, b,
                                                           (short)0, c, false, false);
            }
            float partial = 0.0f;
#pragma unroll
            for (int j = 0; j < 8; ++j) partial += c[j] * wv[j];  // rows h*8+j
            partial += __shfl_xor(partial, 16, 32);               // combine halves
            if (lane < 16) atomicAdd(&outacc[nt * 16 + n], partial);
        }
        __builtin_amdgcn_wave_barrier();
    }
    __syncthreads();
    for (int d = t; d < 256; d += 64)
        out[(size_t)g * 256 + d] = (cnt > 0) ? (outacc[d] + bf2[d]) : 0.0f;
}

// ---------------------------------------------------------------------------
extern "C" void kernel_launch(void* const* d_in, const int* in_sizes, int n_in,
                              void* d_out, int out_size, void* d_ws, size_t ws_size,
                              hipStream_t stream)
{
    const float* x     = (const float*)d_in[0];
    const int*   batch = (const int*)d_in[1];
    const float* wi1   = (const float*)d_in[2];
    const float* bi1   = (const float*)d_in[3];
    const float* gi    = (const float*)d_in[4];
    const float* bti   = (const float*)d_in[5];
    const float* wi2   = (const float*)d_in[6];
    const float* bi2   = (const float*)d_in[7];
    const float* wf1   = (const float*)d_in[8];
    const float* bf1   = (const float*)d_in[9];
    const float* gf    = (const float*)d_in[10];
    const float* btf   = (const float*)d_in[11];
    const float* wf2   = (const float*)d_in[12];
    const float* bf2   = (const float*)d_in[13];
    float* out = (float*)d_out;

    const int N = in_sizes[0] / 256;
    const int B = out_size / 256;

    // workspace layout: 3 packed f16 weights (128 KB each) + imp[N] f32
    _Float16* wi1p = (_Float16*)d_ws;
    _Float16* wf1p = wi1p + 65536;
    _Float16* wf2p = wf1p + 65536;
    float*    imp  = (float*)(wf2p + 65536);

    fr_pack_kernel<<<192, 64, 0, stream>>>(wi1, wf1, wf2, wi1p, wf1p, wf2p);
    fr_imp_kernel<<<(N + 31) / 32, 64, 0, stream>>>(x, wi1p, bi1, gi, bti, wi2, bi2, imp, N);
    fr_readout_kernel<<<B, 64, 0, stream>>>(x, batch, wf1p, wf2p, bf1, gf, btf, bf2, imp, out, N);
}
